// InOutBlock_29025388986817
// MI455X (gfx1250) — matrix operand order repaired and checked by reading, not compiled, not run on hardware
//
#include <hip/hip_runtime.h>
#include <math.h>

// ---------------------------------------------------------------------------
// Problem constants (from reference): B=8, C=56, N=2000, M=200, T=16
// ---------------------------------------------------------------------------
constexpr int Bn  = 8;
constexpr int Cn  = 56;
constexpr int Nn  = 2000;
constexpr int Mn  = 200;
constexpr int Tn  = 16;
constexpr int Gn  = Bn * Cn;        // 448 batched (b,c) GEMMs
constexpr int SPn = Nn * Tn;        // 32000 spatial elements
constexpr int NPn = 2016;           // N padded to multiple of 32 (K padding)
constexpr int MPn = 224;            // M padded to multiple of 32
constexpr int THn = Tn / 2;         // 8 (tconv output length)
constexpr int GB  = 4;              // batches per wave in WMMA kernel
constexpr int WT  = 2;              // w-tiles per wave (B-fragment reuse)

typedef __bf16 bf16;
typedef __attribute__((ext_vector_type(16))) __bf16 v16bf;
typedef __attribute__((ext_vector_type(8)))  __bf16 v8bf;
typedef __attribute__((ext_vector_type(8)))  float  v8f;

union AFrag { v16bf v; v8bf h[2]; };

// ---------------------------------------------------------------------------
// Convert (G, Nin, T) f32 -> (G, T, Npad) bf16 with zero K-padding
// ---------------------------------------------------------------------------
__global__ __launch_bounds__(256)
void k_to_bf16T(const float* __restrict__ x, bf16* __restrict__ xT,
                int Grp, int Nin, int Npad)
{
  long i = (long)blockIdx.x * 256 + threadIdx.x;
  long total = (long)Grp * Tn * Npad;
  if (i >= total) return;
  int  v = (int)(i % Npad);
  int  l = (int)((i / Npad) % Tn);
  long g = i / ((long)Npad * Tn);
  float val = (v < Nin) ? x[(g * Nin + v) * Tn + l] : 0.f;
  xT[i] = (bf16)val;
}

// ---------------------------------------------------------------------------
// Transpose adjacency: A (K, Nout) f32 -> AT (Nout, Kpad) bf16, zero K-pad
// ---------------------------------------------------------------------------
__global__ __launch_bounds__(256)
void k_adjT(const float* __restrict__ A, bf16* __restrict__ AT,
            int K, int Kpad, int Nout)
{
  long i = (long)blockIdx.x * 256 + threadIdx.x;
  long total = (long)Nout * Kpad;
  if (i >= total) return;
  int k = (int)(i % Kpad);
  int w = (int)(i / Kpad);
  float val = (k < K) ? A[(long)k * Nout + w] : 0.f;
  AT[i] = (bf16)val;
}

// ---------------------------------------------------------------------------
// Batched node-contraction GEMM via WMMA bf16 -> f32:
//   y[g, w, l] = sum_v x[g, v, l] * Adj[v, w]
// AT  : (Nout, Kp) bf16  = Adj transposed, K zero-padded
// xT  : (Gtot, T, Kp) bf16 = x transposed, K zero-padded
// yf  : (Gtot, Nout, T) f32   (optional)
// yT  : (Gtot, T, NoutPad) bf16 (optional, for chaining; pad pre-zeroed)
// One wave computes WT 16-row w-tiles x 16(l=T) for GB consecutive g:
// A-fragments reused GB times, B-fragments reused WT times per K-step
// (12 b128 loads per 8 WMMAs in the full-tile path).
// ---------------------------------------------------------------------------
__global__ __launch_bounds__(256)
void k_nconv(const bf16* __restrict__ AT, const bf16* __restrict__ xT,
             float* __restrict__ yf, bf16* __restrict__ yT,
             int Gtot, int Nout, int NoutPad, int Kp)
{
  const int wave = blockIdx.x * (blockDim.x >> 5) + (threadIdx.x >> 5);
  const int lane = threadIdx.x & 31;
  const int nWT  = Nout >> 4;             // 125 w-tiles
  const int nST  = (nWT + WT - 1) / WT;   // 63 super-tiles
  const int nGBk = Gtot / GB;             // 112 g-blocks
  if (wave >= nST * nGBk) return;         // wave-uniform: EXEC stays all-ones
  const int gblk = wave / nST;
  const int st   = wave % nST;
  const int wt0  = st * WT;
  const bool has2 = (wt0 + 1) < nWT;      // last super-tile holds only 1 tile
  const int m    = lane & 15;             // row (A) / column (B) / N-index (D)
  const int sel  = lane >> 4;             // lane half selects K sub-range

  const bf16* arow0 = AT + (size_t)(wt0 * 16 + m) * Kp;
  const bf16* arow1 = arow0 + (size_t)16 * Kp;
  const bf16* brow0 = xT + ((size_t)(gblk * GB) * Tn + m) * Kp + 8 * sel;
  const size_t bstride = (size_t)Tn * Kp;

  v8f acc0[GB] = {};
  v8f acc1[GB] = {};
  if (has2) {
    for (int kb = 0; kb < Kp; kb += 32) {
      AFrag a0, a1;
      // 16-bit A-matrix layout: lanes 0-15 K[0..7]&[16..23], lanes 16-31 +8
      a0.h[0] = *(const v8bf*)(arow0 + kb + 8 * sel);
      a0.h[1] = *(const v8bf*)(arow0 + kb + 16 + 8 * sel);
      a1.h[0] = *(const v8bf*)(arow1 + kb + 8 * sel);
      a1.h[1] = *(const v8bf*)(arow1 + kb + 16 + 8 * sel);
#pragma unroll
      for (int j = 0; j < GB; ++j) {
        // B-matrix layout: lane holds col n=lane%16, k = kb + 8*sel + e (e<8), kb + 16 + 8*sel + (e-8) (e>=8)
        AFrag bb;
        bb.h[0] = *(const v8bf*)(brow0 + (size_t)j * bstride + kb);
        bb.h[1] = *(const v8bf*)(brow0 + (size_t)j * bstride + kb + 16);
        v16bf b = bb.v;
        acc0[j] = __builtin_amdgcn_wmma_f32_16x16x32_bf16(
            false, a0.v, false, b, (short)0, acc0[j], false, false);
        acc1[j] = __builtin_amdgcn_wmma_f32_16x16x32_bf16(
            false, a1.v, false, b, (short)0, acc1[j], false, false);
      }
    }
  } else {
    for (int kb = 0; kb < Kp; kb += 32) {
      AFrag a0;
      a0.h[0] = *(const v8bf*)(arow0 + kb + 8 * sel);
      a0.h[1] = *(const v8bf*)(arow0 + kb + 16 + 8 * sel);
#pragma unroll
      for (int j = 0; j < GB; ++j) {
        AFrag bb;
        bb.h[0] = *(const v8bf*)(brow0 + (size_t)j * bstride + kb);
        bb.h[1] = *(const v8bf*)(brow0 + (size_t)j * bstride + kb + 16);
        v16bf b = bb.v;
        acc0[j] = __builtin_amdgcn_wmma_f32_16x16x32_bf16(
            false, a0.v, false, b, (short)0, acc0[j], false, false);
      }
    }
  }
  // D layout: VGPR r -> row m' = r + 8*sel, col n = lane%16
#pragma unroll 1
  for (int j = 0; j < GB; ++j) {
    const int g = gblk * GB + j;
    if (yf) {
      float* yp = yf + ((size_t)g * Nout + wt0 * 16 + 8 * sel) * Tn + m;
#pragma unroll
      for (int r = 0; r < 8; ++r) yp[(size_t)r * Tn] = acc0[j][r];
      if (has2) {
        float* yq = yp + (size_t)16 * Tn;
#pragma unroll
        for (int r = 0; r < 8; ++r) yq[(size_t)r * Tn] = acc1[j][r];
      }
    }
    if (yT) {
      bf16* tp = yT + ((size_t)g * Tn + m) * NoutPad + wt0 * 16 + 8 * sel;
      v8bf ob;
#pragma unroll
      for (int r = 0; r < 8; ++r) ob[r] = (bf16)acc0[j][r];
      *(v8bf*)tp = ob;
      if (has2) {
#pragma unroll
        for (int r = 0; r < 8; ++r) ob[r] = (bf16)acc1[j][r];
        *(v8bf*)(tp + 16) = ob;
      }
    }
  }
}

// ---------------------------------------------------------------------------
// 1x1 channel conv accumulate: y[b,o,sp] (=|+=) bias[o] + sum_c W[o,woff+c]*x[b,c,sp]
// ---------------------------------------------------------------------------
__global__ __launch_bounds__(256)
void k_c1x1(float* __restrict__ y, const float* __restrict__ x,
            const float* __restrict__ W, int wstride, int woff,
            const float* __restrict__ bias, int mode, int SPtot)
{
  int b  = blockIdx.y;
  int sp = blockIdx.x * 256 + threadIdx.x;
  if (sp >= SPtot) return;
  float xr[Cn];
#pragma unroll
  for (int c = 0; c < Cn; ++c) xr[c] = x[((long)(b * Cn + c)) * SPtot + sp];
#pragma unroll 1
  for (int o = 0; o < Cn; ++o) {
    const float* wr = W + (long)o * wstride + woff;
    float acc = (mode == 0) ? bias[o] : 0.f;
#pragma unroll
    for (int c = 0; c < Cn; ++c) acc = fmaf(wr[c], xr[c], acc);
    long idx = ((long)(b * Cn + o)) * SPtot + sp;
    if (mode == 0) y[idx] = acc; else y[idx] += acc;
  }
}

// ---------------------------------------------------------------------------
// split conv + relu; emits f32 (B,C,M,T) and bf16 transposed (B*C, T, MPn)
// ---------------------------------------------------------------------------
__global__ __launch_bounds__(256)
void k_split(const float* __restrict__ orf, const float* __restrict__ W,
             const float* __restrict__ bias,
             float* __restrict__ of, bf16* __restrict__ oT)
{
  int b  = blockIdx.y;
  int sp = blockIdx.x * 256 + threadIdx.x;
  if (sp >= Mn * Tn) return;
  float xr[Cn];
#pragma unroll
  for (int c = 0; c < Cn; ++c) xr[c] = orf[((long)(b * Cn + c)) * (Mn * Tn) + sp];
  int mm = sp / Tn, tt = sp % Tn;
#pragma unroll 1
  for (int o = 0; o < Cn; ++o) {
    float acc = bias[o];
#pragma unroll
    for (int c = 0; c < Cn; ++c) acc = fmaf(W[o * Cn + c], xr[c], acc);
    acc = fmaxf(acc, 0.f);
    of[((long)(b * Cn + o)) * (Mn * Tn) + sp] = acc;
    oT[((long)(b * Cn + o) * Tn + tt) * MPn + mm] = (bf16)acc;
  }
}

// y = w1[o,v]*(y + resA) + w2[o,v]*(gB + resB)   (residuals folded in)
__global__ __launch_bounds__(256)
void k_combine(float* __restrict__ y, const float* __restrict__ gB,
               const float* __restrict__ resA, const float* __restrict__ resB,
               const float* __restrict__ w1, const float* __restrict__ w2)
{
  long i = (long)blockIdx.x * 256 + threadIdx.x;
  if (i >= (long)Gn * SPn) return;
  int v = (int)((i / Tn) % Nn);
  int o = (int)((i / SPn) % Cn);
  float s1 = w1[o * Nn + v], s2 = w2[o * Nn + v];
  y[i] = s1 * (y[i] + resA[i]) + s2 * (gB[i] + resB[i]);
}

// y += w[o,v] * src
__global__ __launch_bounds__(256)
void k_axpy(float* __restrict__ y, const float* __restrict__ src,
            const float* __restrict__ w)
{
  long i = (long)blockIdx.x * 256 + threadIdx.x;
  if (i >= (long)Gn * SPn) return;
  int v = (int)((i / Tn) % Nn);
  int o = (int)((i / SPn) % Cn);
  y[i] = fmaf(w[o * Nn + v], src[i], y[i]);
}

// ---------------------------------------------------------------------------
// tconv (stride-2 even/odd taps) + tanh*sigmoid gate -> (B,C,N,TH)
// ---------------------------------------------------------------------------
__global__ __launch_bounds__(256)
void k_tconv(const float* __restrict__ x,
             const float* __restrict__ Wf, const float* __restrict__ bf,
             const float* __restrict__ Wg, const float* __restrict__ bg,
             float* __restrict__ out)
{
  __shared__ float lx[Cn][16][Tn];   // 56*16*16*4 = 57 KB of 320 KB LDS
  int b  = blockIdx.y;
  int v0 = blockIdx.x * 16;
  int tid = threadIdx.x;
  int lv = tid / Tn, ll = tid % Tn;
  for (int c = 0; c < Cn; ++c)
    lx[c][lv][ll] = x[(((long)(b * Cn + c)) * Nn + v0 + lv) * Tn + ll];
  __syncthreads();
  for (int t = tid; t < Cn * 16 * THn; t += 256) {
    int h  = t % THn;
    int vv = (t / THn) % 16;
    int o  = t / (THn * 16);
    float f = bf[o], g = bg[o];
#pragma unroll 1
    for (int c = 0; c < Cn; ++c) {
      float xe = lx[c][vv][2 * h], xo = lx[c][vv][2 * h + 1];
      f = fmaf(xe, Wf[(o * Cn + c) * 2 + 0], fmaf(xo, Wf[(o * Cn + c) * 2 + 1], f));
      g = fmaf(xe, Wg[(o * Cn + c) * 2 + 0], fmaf(xo, Wg[(o * Cn + c) * 2 + 1], g));
    }
    float val = tanhf(f) * (1.f / (1.f + expf(-g)));
    out[(((long)(b * Cn + o)) * Nn + v0 + vv) * THn + h] = val;
  }
}

// delta[b,c,h] = sum_v new_out - sum_v new_in ; one block per output
__global__ __launch_bounds__(256)
void k_reduce(const float* __restrict__ nin, const float* __restrict__ nout,
              float* __restrict__ dout)
{
  int idx = blockIdx.x;
  int h  = idx % THn;
  int bc = idx / THn;
  const float* pi = nin  + (long)bc * Nn * THn + h;
  const float* po = nout + (long)bc * Nn * THn + h;
  float s = 0.f;
  for (int v = threadIdx.x; v < Nn; v += 256)
    s += po[(long)v * THn] - pi[(long)v * THn];
  __shared__ float red[256];
  red[threadIdx.x] = s;
  __syncthreads();
  for (int w = 128; w > 0; w >>= 1) {
    if (threadIdx.x < w) red[threadIdx.x] += red[threadIdx.x + w];
    __syncthreads();
  }
  if (threadIdx.x == 0) dout[idx] = red[0];
}

// ---------------------------------------------------------------------------
extern "C" void kernel_launch(void* const* d_in, const int* in_sizes, int n_in,
                              void* d_out, int out_size, void* d_ws, size_t ws_size,
                              hipStream_t stream)
{
  (void)in_sizes; (void)n_in; (void)out_size; (void)ws_size;
  const float* in_data  = (const float*)d_in[0];
  const float* out_data = (const float*)d_in[1];
  const float* orflow   = (const float*)d_in[2];
  const float* adp[6]   = { (const float*)d_in[3], (const float*)d_in[4],  // adp0, adp1
                            (const float*)d_in[5], (const float*)d_in[6],  // xx, yy
                            (const float*)d_in[7], (const float*)d_in[8] };// xy, yx
  const float* orInAdp  = (const float*)d_in[9];
  const float* orOutAdp = (const float*)d_in[10];
  const float* inin_w   = (const float*)d_in[11]; const float* inin_b   = (const float*)d_in[12];
  const float* outout_w = (const float*)d_in[13]; const float* outout_b = (const float*)d_in[14];
  const float* inout_w  = (const float*)d_in[15]; const float* inout_b  = (const float*)d_in[16];
  const float* outin_w  = (const float*)d_in[17]; const float* outin_b  = (const float*)d_in[18];
  const float* orin1_w  = (const float*)d_in[19]; const float* orin1_b  = (const float*)d_in[20];
  const float* orin2_w  = (const float*)d_in[21]; const float* orin2_b  = (const float*)d_in[22];
  const float* orout1_w = (const float*)d_in[23]; const float* orout1_b = (const float*)d_in[24];
  const float* orout2_w = (const float*)d_in[25]; const float* orout2_b = (const float*)d_in[26];
  const float* filt_in_w  = (const float*)d_in[27]; const float* filt_in_b  = (const float*)d_in[28];
  const float* gate_in_w  = (const float*)d_in[29]; const float* gate_in_b  = (const float*)d_in[30];
  const float* filt_out_w = (const float*)d_in[31]; const float* filt_out_b = (const float*)d_in[32];
  const float* gate_out_w = (const float*)d_in[33]; const float* gate_out_b = (const float*)d_in[34];
  const float* split_in_w  = (const float*)d_in[35]; const float* split_in_b  = (const float*)d_in[36];
  const float* split_out_w = (const float*)d_in[37]; const float* split_out_b = (const float*)d_in[38];
  const float* w1_in  = (const float*)d_in[39]; const float* w2_in  = (const float*)d_in[40];
  const float* w3_in  = (const float*)d_in[41]; const float* w4_in  = (const float*)d_in[42];
  const float* w1_out = (const float*)d_in[43]; const float* w2_out = (const float*)d_in[44];
  const float* w3_out = (const float*)d_in[45]; const float* w4_out = (const float*)d_in[46];

  // ---- workspace carve (aligned 256B) ----
  char* p = (char*)d_ws;
  auto carve = [&](size_t bytes) -> char* {
    char* r = p; p += (bytes + 255) & ~(size_t)255; return r;
  };
  bf16* adjT[6];
  for (int i = 0; i < 6; ++i) adjT[i] = (bf16*)carve((size_t)Nn * NPn * 2);
  bf16*  orInT   = (bf16*) carve((size_t)Nn * MPn * 2);
  bf16*  orOutT  = (bf16*) carve((size_t)Nn * MPn * 2);
  bf16*  inT     = (bf16*) carve((size_t)Gn * Tn * NPn * 2);
  bf16*  outT    = (bf16*) carve((size_t)Gn * Tn * NPn * 2);
  float* x1f     = (float*)carve((size_t)Gn * SPn * 4);
  bf16*  x1T     = (bf16*) carve((size_t)Gn * Tn * NPn * 2);
  float* x2f     = (float*)carve((size_t)Gn * SPn * 4);
  float* g_inin  = (float*)carve((size_t)Gn * SPn * 4);
  float* g_outout= (float*)carve((size_t)Gn * SPn * 4);
  float* g_inout = (float*)carve((size_t)Gn * SPn * 4);
  float* g_outin = (float*)carve((size_t)Gn * SPn * 4);
  float* orin_f  = (float*)carve((size_t)Gn * Mn * Tn * 4);
  float* orout_f = (float*)carve((size_t)Gn * Mn * Tn * 4);
  bf16*  orinT   = (bf16*) carve((size_t)Gn * Tn * MPn * 2);
  bf16*  oroutT  = (bf16*) carve((size_t)Gn * Tn * MPn * 2);
  float* orc_y   = (float*)carve((size_t)Gn * SPn * 4);

  auto cdiv = [](long a, long b) { return (int)((a + b - 1) / b); };

  // ---- stage bf16 transposed operands ----
  for (int i = 0; i < 6; ++i)
    k_adjT<<<cdiv((long)Nn * NPn, 256), 256, 0, stream>>>(adp[i], adjT[i], Nn, NPn, Nn);
  k_adjT<<<cdiv((long)Nn * MPn, 256), 256, 0, stream>>>(orInAdp,  orInT,  Mn, MPn, Nn);
  k_adjT<<<cdiv((long)Nn * MPn, 256), 256, 0, stream>>>(orOutAdp, orOutT, Mn, MPn, Nn);
  k_to_bf16T<<<cdiv((long)Gn * Tn * NPn, 256), 256, 0, stream>>>(in_data,  inT,  Gn, Nn, NPn);
  k_to_bf16T<<<cdiv((long)Gn * Tn * NPn, 256), 256, 0, stream>>>(out_data, outT, Gn, Nn, NPn);
  (void)hipMemsetAsync(x1T,    0, (size_t)Gn * Tn * NPn * 2, stream); // K-pad zeros
  (void)hipMemsetAsync(orinT,  0, (size_t)Gn * Tn * MPn * 2, stream);
  (void)hipMemsetAsync(oroutT, 0, (size_t)Gn * Tn * MPn * 2, stream);

  const int nST = ((Nn / 16) + WT - 1) / WT;              // 63 super-tiles
  const int nconvBlocks = (nST * (Gn / GB)) / 8;          // 7056 waves / 8 = 882
  dim3 cgrid(SPn / 256, Bn);                              // (125, 8)
  long tot = (long)Gn * SPn;

  // ---- the four GCNs (WMMA node contractions + incremental 1x1 accumulate)
  auto gcn = [&](const float* xf, const bf16* xT,
                 const bf16* A0, const bf16* A1, const bf16* A2,
                 const float* W, const float* bias, float* y) {
    k_c1x1<<<cgrid, 256, 0, stream>>>(y, xf, W, 7 * Cn, 0, bias, 0, SPn);
    const bf16* As[3] = {A0, A1, A2};
    for (int s = 0; s < 3; ++s) {
      k_nconv<<<nconvBlocks, 256, 0, stream>>>(As[s], xT, x1f, x1T, Gn, Nn, NPn, NPn);
      k_c1x1<<<cgrid, 256, 0, stream>>>(y, x1f, W, 7 * Cn, (2 * s + 1) * Cn, nullptr, 1, SPn);
      k_nconv<<<nconvBlocks, 256, 0, stream>>>(As[s], x1T, x2f, nullptr, Gn, Nn, NPn, NPn);
      k_c1x1<<<cgrid, 256, 0, stream>>>(y, x2f, W, 7 * Cn, (2 * s + 2) * Cn, nullptr, 1, SPn);
    }
  };
  gcn(in_data,  inT,  adjT[0], adjT[1], adjT[2], inin_w,   inin_b,   g_inin);   // adp_xx
  gcn(out_data, outT, adjT[0], adjT[1], adjT[3], outout_w, outout_b, g_outout); // adp_yy
  gcn(in_data,  inT,  adjT[0], adjT[1], adjT[4], inout_w,  inout_b,  g_inout);  // adp_xy
  gcn(out_data, outT, adjT[0], adjT[1], adjT[5], outin_w,  outin_b,  g_outin);  // adp_yx

  // ---- out-region splits (relu 1x1) ----
  dim3 sgrid(cdiv(Mn * Tn, 256), Bn);
  k_split<<<sgrid, 256, 0, stream>>>(orflow, split_in_w,  split_in_b,  orin_f,  orinT);
  k_split<<<sgrid, 256, 0, stream>>>(orflow, split_out_w, split_out_b, orout_f, oroutT);

  // ---- fold residuals + w1/w2 weighting (in-place into g_inin / g_outout)
  k_combine<<<cdiv(tot, 256), 256, 0, stream>>>(g_inin,  g_outin, in_data,  out_data, w1_in,  w2_in);
  k_combine<<<cdiv(tot, 256), 256, 0, stream>>>(g_outout, g_inout, out_data, in_data, w1_out, w2_out);

  // ---- ORC blocks: (M->N WMMA) then (N->N WMMA) then 2-segment 1x1 ----
  auto orc = [&](const bf16* xTs, const bf16* sharedA, const bf16* orA,
                 const float* W, const float* bias) {
    k_nconv<<<nconvBlocks, 256, 0, stream>>>(orA,     xTs, x1f, x1T,    Gn, Nn, NPn, MPn);
    k_nconv<<<nconvBlocks, 256, 0, stream>>>(sharedA, x1T, x2f, nullptr, Gn, Nn, NPn, NPn);
    k_c1x1<<<cgrid, 256, 0, stream>>>(orc_y, x1f, W, 2 * Cn, 0,  bias,    0, SPn);
    k_c1x1<<<cgrid, 256, 0, stream>>>(orc_y, x2f, W, 2 * Cn, Cn, nullptr, 1, SPn);
  };
  orc(orinT,  adjT[2], orInT,  orin1_w,  orin1_b);   // or_in2in
  k_axpy<<<cdiv(tot, 256), 256, 0, stream>>>(g_inin, orc_y, w3_in);
  orc(oroutT, adjT[5], orOutT, orout2_w, orout2_b);  // or_out2in
  k_axpy<<<cdiv(tot, 256), 256, 0, stream>>>(g_inin, orc_y, w4_in);
  orc(oroutT, adjT[3], orOutT, orout1_w, orout1_b);  // or_out2out
  k_axpy<<<cdiv(tot, 256), 256, 0, stream>>>(g_outout, orc_y, w3_out);
  orc(orinT,  adjT[4], orInT,  orin2_w,  orin2_b);   // or_in2out
  k_axpy<<<cdiv(tot, 256), 256, 0, stream>>>(g_outout, orc_y, w4_out);

  // ---- gated temporal convs + node-sum deltas -> d_out ----
  float* outp = (float*)d_out;
  const long OUT1 = (long)Gn * Nn * THn;   // 7,168,000 per tensor
  dim3 tgrid(Nn / 16, Bn);
  k_tconv<<<tgrid, 256, 0, stream>>>(g_inin,   filt_in_w,  filt_in_b,  gate_in_w,  gate_in_b,  outp);
  k_tconv<<<tgrid, 256, 0, stream>>>(g_outout, filt_out_w, filt_out_b, gate_out_w, gate_out_b, outp + OUT1);
  k_reduce<<<Gn * THn, 256, 0, stream>>>(outp, outp + OUT1, outp + 2 * OUT1);
}
